// Topn_LF_DNN_52845277610547
// MI455X (gfx1250) — matrix operand order repaired and checked by reading, not compiled, not run on hardware
//
#include <hip/hip_runtime.h>
#include <stdint.h>

// ---------------------------------------------------------------------------
// CDNA5 / gfx1250 (wave32). WMMA bf16 16x16x32, f32 accumulate.
// ---------------------------------------------------------------------------
typedef __attribute__((ext_vector_type(16))) __bf16 v16bf;
typedef __attribute__((ext_vector_type(8)))  float  v8f;

#define NB    16384
#define TDIM  768
#define ADIM  512
#define VDIM  768
#define EDIM  512
#define KTOT  2048   // TDIM+ADIM+VDIM
#define H1D   512
#define H2D   128
#define APAD  48     // LDS row pitch in bf16 elements (96B, 16B aligned)

__device__ __forceinline__ uint16_t f2bf(float f) {
  uint32_t u = __float_as_uint(f);
  return (uint16_t)((u + 0x7FFFu + ((u >> 16) & 1u)) >> 16);  // RNE
}

union FragBF { uint4 q[2]; v16bf v; };

// ---------------------------------------------------------------------------
// Small generic fp32 GEMM for one-time weight folding: C (+)= A[MxK] @ B[KxN]
// All M,N,K multiples of 16 in every use.
// ---------------------------------------------------------------------------
__global__ void gemm_f32(const float* __restrict__ A, const float* __restrict__ Bm,
                         float* __restrict__ C, int M, int N, int K, int accumulate) {
  __shared__ float As[16][17];
  __shared__ float Bs[16][17];
  const int tx = threadIdx.x, ty = threadIdx.y;
  const int col = blockIdx.x * 16 + tx;
  const int row = blockIdx.y * 16 + ty;
  float acc = 0.f;
  for (int k0 = 0; k0 < K; k0 += 16) {
    As[ty][tx] = A[(size_t)row * K + k0 + tx];
    Bs[ty][tx] = Bm[(size_t)(k0 + ty) * N + col];
    __syncthreads();
#pragma unroll
    for (int kk = 0; kk < 16; ++kk) acc += As[ty][kk] * Bs[kk][tx];
    __syncthreads();
  }
  if (accumulate) C[(size_t)row * N + col] += acc;
  else            C[(size_t)row * N + col]  = acc;
}

// dst[n*K+k] = bf16(src[k*N+n])
__global__ void transpose_bf16(const float* __restrict__ src, uint16_t* __restrict__ dst,
                               int K, int N) {
  int k = blockIdx.x * 16 + threadIdx.x;
  int n = blockIdx.y * 16 + threadIdx.y;
  if (k < K && n < N) dst[(size_t)n * K + k] = f2bf(src[(size_t)k * N + n]);
}

// bvo[n] = sum_k bv[k]*Wo[k,n] + bo[n]      (512 threads)
__global__ void bias_vo(const float* __restrict__ bv, const float* __restrict__ Wo,
                        const float* __restrict__ bo, float* __restrict__ out) {
  int n = threadIdx.x;
  float s = bo[n];
  for (int k = 0; k < EDIM; ++k) s += bv[k] * Wo[(size_t)k * EDIM + n];
  out[n] = s;
}

// bc[n] = b1[n] + sum_p sum_k bvo_p[k] * W1[p*512+k, n]   (512 threads)
__global__ void bias_comb(const float* __restrict__ b1, const float* __restrict__ W1,
                          const float* __restrict__ v0, const float* __restrict__ v1,
                          const float* __restrict__ v2, const float* __restrict__ v3,
                          const float* __restrict__ v4, float* __restrict__ out) {
  int n = threadIdx.x;
  const float* vs[5] = {v0, v1, v2, v3, v4};
  float s = b1[n];
  for (int p = 0; p < 5; ++p) {
    const float* bv = vs[p];
    const float* Wp = W1 + (size_t)p * EDIM * H1D;
    float t = 0.f;
    for (int k = 0; k < EDIM; ++k) t += bv[k] * Wp[(size_t)k * H1D + n];
    s += t;
  }
  out[n] = s;
}

// ---------------------------------------------------------------------------
// Router: one wave (32 lanes) per sample; 3 dot products + argmax.
// ---------------------------------------------------------------------------
__global__ void zero_counts(int* c) { if (threadIdx.x < 3) c[threadIdx.x] = 0; }

__global__ void __launch_bounds__(256)
logits_argmax(const float* __restrict__ texts, const float* __restrict__ audios,
              const float* __restrict__ videos,
              const float* __restrict__ wt, const float* __restrict__ wa,
              const float* __restrict__ wvv,
              const float* __restrict__ bt, const float* __restrict__ ba,
              const float* __restrict__ bvb,
              int* __restrict__ idx, int* __restrict__ counts) {
  int wave = (blockIdx.x * blockDim.x + threadIdx.x) >> 5;
  int lane = threadIdx.x & 31;
  if (wave >= NB) return;
  const float4* t4 = (const float4*)(texts  + (size_t)wave * TDIM);
  const float4* a4 = (const float4*)(audios + (size_t)wave * ADIM);
  const float4* v4 = (const float4*)(videos + (size_t)wave * VDIM);
  const float4* wt4 = (const float4*)wt;
  const float4* wa4 = (const float4*)wa;
  const float4* wv4 = (const float4*)wvv;
  float s0 = 0.f, s1 = 0.f, s2 = 0.f;
  for (int k = lane; k < TDIM / 4; k += 32) { float4 a = t4[k], w = wt4[k]; s0 += a.x*w.x + a.y*w.y + a.z*w.z + a.w*w.w; }
  for (int k = lane; k < ADIM / 4; k += 32) { float4 a = a4[k], w = wa4[k]; s1 += a.x*w.x + a.y*w.y + a.z*w.z + a.w*w.w; }
  for (int k = lane; k < VDIM / 4; k += 32) { float4 a = v4[k], w = wv4[k]; s2 += a.x*w.x + a.y*w.y + a.z*w.z + a.w*w.w; }
#pragma unroll
  for (int off = 16; off; off >>= 1) {
    s0 += __shfl_xor(s0, off, 32);
    s1 += __shfl_xor(s1, off, 32);
    s2 += __shfl_xor(s2, off, 32);
  }
  if (lane == 0) {
    float l0 = s0 + bt[0], l1 = s1 + ba[0], l2 = s2 + bvb[0];
    int c = 0; float best = l0;
    if (l1 > best) { best = l1; c = 1; }
    if (l2 > best) { c = 2; }
    idx[wave] = c;
    atomicAdd(&counts[c], 1);
  }
}

__global__ void make_bases(const int* __restrict__ counts, int* __restrict__ base,
                           int* __restrict__ cursor) {
  base[0] = 0; base[1] = counts[0]; base[2] = counts[0] + counts[1];
  cursor[0] = base[0]; cursor[1] = base[1]; cursor[2] = base[2];
}

__global__ void build_perm(const int* __restrict__ idx, int* __restrict__ cursor,
                           int* __restrict__ perm) {
  int i = blockIdx.x * blockDim.x + threadIdx.x;
  if (i >= NB) return;
  int pos = atomicAdd(&cursor[idx[i]], 1);
  perm[pos] = i;
}

// ---------------------------------------------------------------------------
// Main fused GEMM: h1c[g, :] = relu( concat(T,A,V)[perm[g], :] @ Wcomb[c] + bc[c] )
// Block = 256 thr (8 waves), tile 128(M) x 128(N), K-chunk 32, bf16 WMMA.
// grid.x = 4 N-tiles, grid.y = 3 buckets * 128 row-tiles.
// ---------------------------------------------------------------------------
__global__ void __launch_bounds__(256)
gemm1_wmma(const float* __restrict__ texts, const float* __restrict__ audios,
           const float* __restrict__ videos,
           const uint16_t* __restrict__ WcombT,   // [3][512][2048] bf16 (N-major)
           const float* __restrict__ bc,          // [3][512]
           const int* __restrict__ base, const int* __restrict__ counts,
           const int* __restrict__ perm,
           uint16_t* __restrict__ h1c)            // [NB][512] bf16 compact
{
  __shared__ uint16_t As[128 * APAD];
  __shared__ uint16_t Bs[128 * APAD];
  __shared__ int rowIdx[128];

  const int ntile = blockIdx.x;
  const int c = blockIdx.y >> 7;
  const int t = blockIdx.y & 127;
  const int cnt = counts[c];
  const int rb  = base[c] + t * 128;
  const int nrows = cnt - t * 128;
  if (nrows <= 0) return;

  const int tid  = threadIdx.x;
  if (tid < 128) rowIdx[tid] = perm[rb + (tid < nrows ? tid : 0)];
  __syncthreads();

  const int lane = tid & 31;
  const int wid  = tid >> 5;
  const int wm   = wid >> 1;           // 0..3 -> 32-row slice
  const int wn   = wid & 1;            // 0..1 -> 64-col slice

  v8f acc[2][4];
#pragma unroll
  for (int i = 0; i < 2; ++i)
#pragma unroll
    for (int j = 0; j < 4; ++j) acc[i][j] = (v8f){0.f,0.f,0.f,0.f,0.f,0.f,0.f,0.f};

  const uint16_t* Wc = WcombT + (size_t)c * 512 * KTOT + (size_t)ntile * 128 * KTOT;
  const int lm = tid >> 1;             // 0..127 staging row
  const int lk = (tid & 1) * 16;       // 0 or 16

  for (int kc = 0; kc < KTOT / 32; ++kc) {
    const int kg = kc * 32;
    const float* src; int cols, koff;
    if (kg < TDIM)             { src = texts;  cols = TDIM; koff = kg; }
    else if (kg < TDIM + ADIM) { src = audios; cols = ADIM; koff = kg - TDIM; }
    else                       { src = videos; cols = VDIM; koff = kg - TDIM - ADIM; }

    // --- stage A (gathered rows, fp32 -> bf16), pack in regs, 16B LDS stores
    {
      const float4* p4 = (const float4*)(src + (size_t)rowIdx[lm] * cols + koff + lk);
      uint32_t pk[8];
#pragma unroll
      for (int q = 0; q < 4; ++q) {
        float4 f = p4[q];
        pk[2*q]   = (uint32_t)f2bf(f.x) | ((uint32_t)f2bf(f.y) << 16);
        pk[2*q+1] = (uint32_t)f2bf(f.z) | ((uint32_t)f2bf(f.w) << 16);
      }
      *(uint4*)&As[lm * APAD + lk]     = *(const uint4*)&pk[0];
      *(uint4*)&As[lm * APAD + lk + 8] = *(const uint4*)&pk[4];
    }
    // --- stage B (weights, already bf16 N-major: contiguous K)
    {
      const uint4* p = (const uint4*)(Wc + (size_t)lm * KTOT + kg + lk);
      *(uint4*)&Bs[lm * APAD + lk]     = p[0];
      *(uint4*)&Bs[lm * APAD + lk + 8] = p[1];
      if (kc + 1 < KTOT / 32)
        __builtin_prefetch(Wc + (size_t)lm * KTOT + kg + 32 + lk, 0, 3);
    }
    __syncthreads();

    // --- A fragments: lane<16 holds K [0,8)+[16,24); lane>=16 holds [8,16)+[24,32)
    const int arow = wm * 32 + (lane & 15);
    const int akb  = (lane >> 4) * 8;
    FragBF a0, a1;
    a0.q[0] = *(const uint4*)&As[arow * APAD + akb];
    a0.q[1] = *(const uint4*)&As[arow * APAD + akb + 16];
    a1.q[0] = *(const uint4*)&As[(arow + 16) * APAD + akb];
    a1.q[1] = *(const uint4*)&As[(arow + 16) * APAD + akb + 16];
    const int bkb = (lane >> 4) * 8;
#pragma unroll
    for (int j = 0; j < 4; ++j) {
      const int bn = wn * 64 + j * 16 + (lane & 15);
      FragBF b;
      b.q[0] = *(const uint4*)&Bs[bn * APAD + bkb];
      b.q[1] = *(const uint4*)&Bs[bn * APAD + bkb + 16];
      acc[0][j] = __builtin_amdgcn_wmma_f32_16x16x32_bf16(false, a0.v, false, b.v,
                                                          (short)0, acc[0][j], false, false);
      acc[1][j] = __builtin_amdgcn_wmma_f32_16x16x32_bf16(false, a1.v, false, b.v,
                                                          (short)0, acc[1][j], false, false);
    }
    __syncthreads();
  }

  // --- epilogue: bias + relu -> compact bf16 h1
  const float* bcc = bc + c * 512 + ntile * 128;
#pragma unroll
  for (int i = 0; i < 2; ++i) {
    const int mrow = wm * 32 + i * 16 + (lane >> 4) * 8;
#pragma unroll
    for (int j = 0; j < 4; ++j) {
      const int n = wn * 64 + j * 16 + (lane & 15);
      const float bias = bcc[n];
#pragma unroll
      for (int v = 0; v < 8; ++v) {
        const int m = mrow + v;
        if (m < nrows) {
          float x = acc[i][j][v] + bias;
          x = x > 0.f ? x : 0.f;
          h1c[(size_t)(rb + m) * H1D + ntile * 128 + n] = f2bf(x);
        }
      }
    }
  }
}

// ---------------------------------------------------------------------------
// Fused tail: h2 = relu(h1c @ W2 + b2); out[perm[g]] = h2 . W3 + b3
// Block = 256 thr, tile 128(M) x 128(N=full), K = 512.
// ---------------------------------------------------------------------------
__global__ void __launch_bounds__(256)
gemm2_out(const uint16_t* __restrict__ h1c, const uint16_t* __restrict__ W2T, // [128][512]
          const float* __restrict__ b2, const float* __restrict__ W3,
          const float* __restrict__ b3, const int* __restrict__ perm,
          float* __restrict__ out) {
  __shared__ uint16_t As[128 * APAD];
  __shared__ uint16_t Bs[128 * APAD];
  __shared__ float h2s[128][H2D + 4];

  const int t   = blockIdx.x;
  const int tid = threadIdx.x;
  const int lane = tid & 31;
  const int wid = tid >> 5;
  const int wm = wid >> 1, wn = wid & 1;
  const int lm = tid >> 1, lk = (tid & 1) * 16;

  v8f acc[2][4];
#pragma unroll
  for (int i = 0; i < 2; ++i)
#pragma unroll
    for (int j = 0; j < 4; ++j) acc[i][j] = (v8f){0.f,0.f,0.f,0.f,0.f,0.f,0.f,0.f};

  for (int kc = 0; kc < H1D / 32; ++kc) {
    const int kg = kc * 32;
    const uint4* pa = (const uint4*)(h1c + (size_t)(t * 128 + lm) * H1D + kg + lk);
    *(uint4*)&As[lm * APAD + lk]     = pa[0];
    *(uint4*)&As[lm * APAD + lk + 8] = pa[1];
    const uint4* pb = (const uint4*)(W2T + (size_t)lm * H1D + kg + lk);
    *(uint4*)&Bs[lm * APAD + lk]     = pb[0];
    *(uint4*)&Bs[lm * APAD + lk + 8] = pb[1];
    __syncthreads();

    const int arow = wm * 32 + (lane & 15);
    const int akb  = (lane >> 4) * 8;
    FragBF a0, a1;
    a0.q[0] = *(const uint4*)&As[arow * APAD + akb];
    a0.q[1] = *(const uint4*)&As[arow * APAD + akb + 16];
    a1.q[0] = *(const uint4*)&As[(arow + 16) * APAD + akb];
    a1.q[1] = *(const uint4*)&As[(arow + 16) * APAD + akb + 16];
    const int bkb = (lane >> 4) * 8;
#pragma unroll
    for (int j = 0; j < 4; ++j) {
      const int bn = wn * 64 + j * 16 + (lane & 15);
      FragBF b;
      b.q[0] = *(const uint4*)&Bs[bn * APAD + bkb];
      b.q[1] = *(const uint4*)&Bs[bn * APAD + bkb + 16];
      acc[0][j] = __builtin_amdgcn_wmma_f32_16x16x32_bf16(false, a0.v, false, b.v,
                                                          (short)0, acc[0][j], false, false);
      acc[1][j] = __builtin_amdgcn_wmma_f32_16x16x32_bf16(false, a1.v, false, b.v,
                                                          (short)0, acc[1][j], false, false);
    }
    __syncthreads();
  }

  // bias + relu into LDS
#pragma unroll
  for (int i = 0; i < 2; ++i) {
    const int mrow = wm * 32 + i * 16 + (lane >> 4) * 8;
#pragma unroll
    for (int j = 0; j < 4; ++j) {
      const int n = wn * 64 + j * 16 + (lane & 15);
      const float bias = b2[n];
#pragma unroll
      for (int v = 0; v < 8; ++v) {
        float x = acc[i][j][v] + bias;
        h2s[mrow + v][n] = x > 0.f ? x : 0.f;
      }
    }
  }
  __syncthreads();

  if (tid < 128) {
    float s = 0.f;
#pragma unroll 8
    for (int k = 0; k < H2D; ++k) s += h2s[tid][k] * W3[k];
    out[perm[t * 128 + tid]] = s + b3[0];
  }
}

// ---------------------------------------------------------------------------
// Host orchestration
// ---------------------------------------------------------------------------
extern "C" void kernel_launch(void* const* d_in, const int* in_sizes, int n_in,
                              void* d_out, int out_size, void* d_ws, size_t ws_size,
                              hipStream_t stream) {
  (void)in_sizes; (void)n_in; (void)out_size; (void)ws_size;

  // Flattening assumption: setup_inputs() dict order, params nested depth-first
  // in insertion order. Each _mha_params group: Wq,bq,Wk,bk,Wv,bv,Wo,bo.
  const float* texts  = (const float*)d_in[0];
  const float* audios = (const float*)d_in[1];
  const float* videos = (const float*)d_in[2];
  const float* wt  = (const float*)d_in[3];
  const float* wa  = (const float*)d_in[4];
  const float* wvv = (const float*)d_in[5];
  const float* bt  = (const float*)d_in[6];
  const float* ba  = (const float*)d_in[7];
  const float* bvb = (const float*)d_in[8];
  // Feature order: 0=self_t 1=self_a 2=self_v 3=TA 4=TV 5=VA 6=VT 7=AT 8=AV
  static const int grp[9]  = {9, 17, 25, 33, 41, 49, 57, 65, 73};
  static const int fdim[9] = {768, 512, 768, 512, 768, 512, 768, 768, 768};
  const float* W1 = (const float*)d_in[81];
  const float* b1 = (const float*)d_in[82];
  const float* W2 = (const float*)d_in[83];
  const float* b2 = (const float*)d_in[84];
  const float* W3 = (const float*)d_in[85];
  const float* b3 = (const float*)d_in[86];
  float* out = (float*)d_out;

  // --- workspace layout (256B aligned blobs) ---
  uintptr_t wsp = (uintptr_t)d_ws;
  size_t off = 0;
  auto alloc = [&](size_t bytes) -> void* {
    void* p = (void*)(wsp + off);
    off = (off + bytes + 255) & ~(size_t)255;
    return p;
  };
  size_t wvoOff[10]; wvoOff[0] = 0;
  for (int f = 0; f < 9; ++f) wvoOff[f + 1] = wvoOff[f] + (size_t)fdim[f] * EDIM;
  float*    Wvo    = (float*)   alloc(wvoOff[9] * 4);                 // 12.6 MB
  float*    Wcomb  = (float*)   alloc((size_t)3 * KTOT * H1D * 4);    // 12.6 MB
  uint16_t* WcombT = (uint16_t*)alloc((size_t)3 * H1D * KTOT * 2);    //  6.3 MB
  uint16_t* W2T    = (uint16_t*)alloc((size_t)H2D * H1D * 2);
  float*    bvo    = (float*)   alloc((size_t)9 * EDIM * 4);
  float*    bc     = (float*)   alloc((size_t)3 * H1D * 4);
  int*      idx    = (int*)     alloc((size_t)NB * 4);
  int*      perm   = (int*)     alloc((size_t)NB * 4);
  int*      counts = (int*)     alloc(3 * 4);
  int*      bases  = (int*)     alloc(3 * 4);
  int*      cursor = (int*)     alloc(3 * 4);
  uint16_t* h1c    = (uint16_t*)alloc((size_t)NB * H1D * 2);          // 16.8 MB

  dim3 blk16(16, 16);

  // P1: Wvo_f = Wv_f @ Wo_f   (9 small GEMMs)
  for (int f = 0; f < 9; ++f) {
    const float* Wv = (const float*)d_in[grp[f] + 4];
    const float* Wo = (const float*)d_in[grp[f] + 6];
    gemm_f32<<<dim3(EDIM / 16, fdim[f] / 16), blk16, 0, stream>>>(
        Wv, Wo, Wvo + wvoOff[f], fdim[f], EDIM, EDIM, 0);
  }

  // P2: Wcomb[c] region += Wvo_f @ W1_p   (15 small GEMMs; beta-0 first per region)
  struct P2 { int c, f, p, rowoff, beta; };
  static const P2 p2tab[15] = {
      {0, 0, 0, 0, 0},    {0, 7, 2, 0, 1},    {0, 6, 4, 0, 1},
      {0, 3, 1, 768, 0},  {0, 4, 3, 1280, 0},
      {1, 7, 1, 0, 0},    {1, 1, 0, 768, 0},  {1, 3, 2, 768, 1},
      {1, 5, 4, 768, 1},  {1, 8, 3, 1280, 0},
      {2, 6, 3, 0, 0},    {2, 5, 1, 768, 0},  {2, 2, 0, 1280, 0},
      {2, 8, 2, 1280, 1}, {2, 4, 4, 1280, 1},
  };
  for (int i = 0; i < 15; ++i) {
    const P2& e = p2tab[i];
    gemm_f32<<<dim3(H1D / 16, fdim[e.f] / 16), blk16, 0, stream>>>(
        Wvo + wvoOff[e.f], W1 + (size_t)e.p * EDIM * H1D,
        Wcomb + (size_t)e.c * KTOT * H1D + (size_t)e.rowoff * H1D,
        fdim[e.f], H1D, EDIM, e.beta);
  }

  // P3: bf16 N-major weight copies
  for (int c = 0; c < 3; ++c)
    transpose_bf16<<<dim3(KTOT / 16, H1D / 16), blk16, 0, stream>>>(
        Wcomb + (size_t)c * KTOT * H1D, WcombT + (size_t)c * H1D * KTOT, KTOT, H1D);
  transpose_bf16<<<dim3(H1D / 16, H2D / 16), blk16, 0, stream>>>(W2, W2T, H1D, H2D);

  // P4: fold biases (all zero in the reference setup, but computed for generality)
  for (int f = 0; f < 9; ++f)
    bias_vo<<<1, EDIM, 0, stream>>>((const float*)d_in[grp[f] + 5],
                                    (const float*)d_in[grp[f] + 6],
                                    (const float*)d_in[grp[f] + 7], bvo + f * EDIM);
  static const int bcf[3][5] = {{0, 3, 7, 4, 6}, {1, 7, 3, 8, 5}, {2, 5, 8, 6, 4}};
  for (int c = 0; c < 3; ++c)
    bias_comb<<<1, H1D, 0, stream>>>(b1, W1,
                                     bvo + bcf[c][0] * EDIM, bvo + bcf[c][1] * EDIM,
                                     bvo + bcf[c][2] * EDIM, bvo + bcf[c][3] * EDIM,
                                     bvo + bcf[c][4] * EDIM, bc + c * H1D);

  // Router: idx + class bucketing (re-zeroed every call -> graph-replay safe)
  zero_counts<<<1, 32, 0, stream>>>(counts);
  logits_argmax<<<NB * 32 / 256, 256, 0, stream>>>(texts, audios, videos,
                                                   wt, wa, wvv, bt, ba, bvb,
                                                   idx, counts);
  make_bases<<<1, 1, 0, stream>>>(counts, bases, cursor);
  build_perm<<<NB / 256, 256, 0, stream>>>(idx, cursor, perm);

  // Main WMMA GEMM + fused tail
  gemm1_wmma<<<dim3(4, 3 * (NB / 128)), 256, 0, stream>>>(
      texts, audios, videos, WcombT, bc, bases, counts, perm, h1c);
  gemm2_out<<<NB / 128, 256, 0, stream>>>(h1c, W2T, b2, W3, b3, perm, out);
}
